// VMambaCrossScanBlock_30348238913548
// MI455X (gfx1250) — compile-verified, matrix operand order checked by reading, not run on hardware
//
#include <hip/hip_runtime.h>
#include <hip/hip_bf16.h>
#include <math.h>

// ---------------------------------------------------------------------------
// VMamba cross-scan block for MI455X (gfx1250, wave32, WMMA bf16)
// B=4 H=32 W=32 D=128 N=48 R=24 K=3
// ---------------------------------------------------------------------------

typedef __bf16 bf16_t;
typedef __attribute__((ext_vector_type(16))) __bf16 v16bf;
typedef __attribute__((ext_vector_type(8)))  float  v8f;

#define DEVINL static __device__ __forceinline__

DEVINL float clampf(float x, float lo, float hi) { return fminf(fmaxf(x, lo), hi); }

DEVINL float nan_fix(float v) {
    if (isnan(v)) return 0.0f;
    if (isinf(v)) return v > 0.0f ? 1.0f : -1.0f;
    return v;
}

// WMMA fragment loader. For D = A(16x32) x B(32x16) + C in bf16:
//   A: lane L holds row (L&15), K-chunks [h*8 .. h*8+7] and [h*8+16 .. h*8+23], h=L>>4
//   B (= W.T with W row-major, K contiguous): identical addressing on rows of W.
template <typename P>
DEVINL v16bf frag_ld(P p, int stride, int lane) {
    const int r = lane & 15, hh = lane >> 4;
    p += r * stride + hh * 8;
    v16bf f;
#pragma unroll
    for (int i = 0; i < 8; ++i) { f[i] = p[i]; f[8 + i] = p[16 + i]; }
    return f;
}

DEVINL v8f wmma_bf16(v16bf a, v16bf b, v8f c) {
    return __builtin_amdgcn_wmma_f32_16x16x32_bf16(false, a, false, b, (short)0, c,
                                                   false, false);
}

// ---------------------------------------------------------------------------
// Weight prep: f32 -> bf16 (plus zero-padded variants)
// ---------------------------------------------------------------------------
__global__ void k_cvt(const float* __restrict__ s, bf16_t* __restrict__ d, int n) {
    for (int i = blockIdx.x * blockDim.x + threadIdx.x; i < n; i += gridDim.x * blockDim.x)
        d[i] = (bf16_t)s[i];
}

// sWx (4,120,128) -> (4,128,128), rows >=120 zero
__global__ void k_padwx(const float* __restrict__ s, bf16_t* __restrict__ d) {
    int i = blockIdx.x * blockDim.x + threadIdx.x;
    if (i >= 4 * 128 * 128) return;
    int c = i & 127, r = (i >> 7) & 127, g = i >> 14;
    d[i] = (r < 120) ? (bf16_t)s[(g * 120 + r) * 128 + c] : (bf16_t)0.0f;
}

// sWdt (4,128,24) -> (4,128,32), cols >=24 zero
__global__ void k_padwdt(const float* __restrict__ s, bf16_t* __restrict__ d) {
    int i = blockIdx.x * blockDim.x + threadIdx.x;
    if (i >= 4 * 128 * 32) return;
    int c = i & 31, r = (i >> 5) & 127, g = i >> 12;
    d[i] = (c < 24) ? (bf16_t)s[(g * 128 + r) * 24 + c] : (bf16_t)0.0f;
}

// ---------------------------------------------------------------------------
// Kernel 1: x = LN(grid)*ng+nb @ Win.T + bin   (M=4096, N=128, K=128)
// 256 blocks x 128 threads (16-row tile per block, 4 waves, 2 N-tiles/wave)
// ---------------------------------------------------------------------------
__global__ void __launch_bounds__(128) k_ln_proj(
    const float* __restrict__ grid, const float* __restrict__ ng,
    const float* __restrict__ nb, const bf16_t* __restrict__ Winb,
    const float* __restrict__ bin, float* __restrict__ xout) {
    __shared__ bf16_t As[16][136];
    const int tile = blockIdx.x, tid = threadIdx.x;
    const int row = tid >> 3, sub = tid & 7;  // 8 lanes per row (within a wave32)
    const float* g = grid + (tile * 16 + row) * 128 + sub * 16;
    float v[16], s = 0.f, s2 = 0.f;
#pragma unroll
    for (int i = 0; i < 16; ++i) { v[i] = g[i]; s += v[i]; s2 += v[i] * v[i]; }
#pragma unroll
    for (int m = 4; m >= 1; m >>= 1) { s += __shfl_xor(s, m, 8); s2 += __shfl_xor(s2, m, 8); }
    const float mean = s * (1.f / 128.f);
    const float rstd = rsqrtf(s2 * (1.f / 128.f) - mean * mean + 1e-5f);
#pragma unroll
    for (int i = 0; i < 16; ++i) {
        int c = sub * 16 + i;
        As[row][c] = (bf16_t)((v[i] - mean) * rstd * ng[c] + nb[c]);
    }
    __syncthreads();
    const int wave = tid >> 5, lane = tid & 31;
#pragma unroll
    for (int j = 0; j < 2; ++j) {
        const int n0 = wave * 32 + j * 16;
        v8f acc = {};
#pragma unroll
        for (int kk = 0; kk < 128; kk += 32) {
            v16bf a = frag_ld(&As[0][0] + kk, 136, lane);
            v16bf b = frag_ld(Winb + n0 * 128 + kk, 128, lane);
            acc = wmma_bf16(a, b, acc);
        }
        const int n = n0 + (lane & 15), mb = (lane >> 4) * 8;
#pragma unroll
        for (int i = 0; i < 8; ++i)
            xout[(tile * 16 + mb + i) * 128 + n] = acc[i] + bin[n];
    }
}

// ---------------------------------------------------------------------------
// Kernel 2: depthwise 3x3 conv + bias, mask by valid; write xm (BHWD) and
// transposed xmT (BWHD) for the column scans.
// ---------------------------------------------------------------------------
__global__ void k_conv(const float* __restrict__ x, const float* __restrict__ convw,
                       const float* __restrict__ convb,
                       const unsigned char* __restrict__ valid,
                       float* __restrict__ xm, float* __restrict__ xmT) {
    const int pix = blockIdx.x;   // 4096
    const int d = threadIdx.x;    // 128
    const int w = pix & 31, h = (pix >> 5) & 31, b = pix >> 10;
    float acc = convb[d];
#pragma unroll
    for (int kh = 0; kh < 3; ++kh) {
        int hh = h + kh - 1;
        if (hh < 0 || hh > 31) continue;
#pragma unroll
        for (int kw = 0; kw < 3; ++kw) {
            int ww = w + kw - 1;
            if (ww < 0 || ww > 31) continue;
            acc += x[((b * 32 + hh) * 32 + ww) * 128 + d] * convw[(kh * 3 + kw) * 128 + d];
        }
    }
    acc *= valid[pix] ? 1.f : 0.f;
    xm[pix * 128 + d] = acc;
    xmT[((b * 32 + w) * 32 + h) * 128 + d] = acc;
}

// ---------------------------------------------------------------------------
// Kernel 3: fused per-sequence selective scan.
// 512 blocks = 4 dirs x 128 sequences; 128 threads (4 wave32s).
// LDS buffers aliased across pipeline stages (50688 B total).
// ---------------------------------------------------------------------------
__global__ void __launch_bounds__(128) k_scan(
    const float* __restrict__ xm, const float* __restrict__ xmT,
    const bf16_t* __restrict__ sWinb, const bf16_t* __restrict__ sWxb,
    const bf16_t* __restrict__ sWdtb, const float* __restrict__ sbdt,
    const float* __restrict__ sAlog, const float* __restrict__ sD,
    const bf16_t* __restrict__ sWoutb, const float* __restrict__ sng,
    const float* __restrict__ snb, float* __restrict__ dir_out) {
    __shared__ __align__(16) char smem[50688];
    float*  uS  = (float*)(smem);              // 32x128 f32; later reused as Yo
    bf16_t* zbf = (bf16_t*)(smem + 16384);     // 32x128 bf16 (z gate)
    bf16_t* AsP = (bf16_t*)(smem + 24576);     // As 32x136 -> pbf 32x128
    bf16_t* UbD = (bf16_t*)(smem + 33280);     // Ub 32x136 -> delta 32x128
    bf16_t* dtY = (bf16_t*)(smem + 41984);     // dtA 32x40 -> Yb 32x136

    const int dir = blockIdx.x >> 7;
    const int seq = blockIdx.x & 127;
    const int tid = threadIdx.x, wave = tid >> 5, lane = tid & 31;
    const bool flip = dir & 1;
    const float* xbase = ((dir < 2) ? xm : xmT) + seq * 32 * 128;

    // stage x tile (flip in sequence dim) as bf16 A-matrix
    bf16_t* As = AsP;
    for (int idx = tid; idx < 32 * 128; idx += 128) {
        int l = idx >> 7, c = idx & 127;
        int ls = flip ? 31 - l : l;
        As[l * 136 + c] = (bf16_t)xbase[ls * 128 + c];
    }
    __syncthreads();

    // GEMM1: xz(32x256) = x @ sWin[dir].T ; split u (f32+bf16) / z (bf16)
    const bf16_t* Wi = sWinb + dir * 256 * 128;
    bf16_t* Ub = UbD;
#pragma unroll
    for (int mt = 0; mt < 2; ++mt)
#pragma unroll
        for (int j = 0; j < 4; ++j) {
            int n0 = wave * 64 + j * 16;
            v8f acc = {};
#pragma unroll
            for (int kk = 0; kk < 128; kk += 32) {
                v16bf a = frag_ld(As + mt * 16 * 136 + kk, 136, lane);
                v16bf b = frag_ld(Wi + n0 * 128 + kk, 128, lane);
                acc = wmma_bf16(a, b, acc);
            }
            int n = n0 + (lane & 15), mb = mt * 16 + (lane >> 4) * 8;
#pragma unroll
            for (int i = 0; i < 8; ++i) {
                int m = mb + i;
                float v = acc[i];
                if (n < 128) { uS[m * 128 + n] = v; Ub[m * 136 + n] = (bf16_t)v; }
                else         { zbf[m * 128 + (n - 128)] = (bf16_t)v; }
            }
        }
    __syncthreads();

    // GEMM2: p(32x128pad) = u @ sWxPad[dir].T  (cols 0..119 valid)
    const bf16_t* Wx = sWxb + dir * 128 * 128;
    bf16_t* pbf = AsP;  // reuse (As dead)
#pragma unroll
    for (int mt = 0; mt < 2; ++mt)
#pragma unroll
        for (int j = 0; j < 2; ++j) {
            int n0 = wave * 32 + j * 16;
            v8f acc = {};
#pragma unroll
            for (int kk = 0; kk < 128; kk += 32) {
                v16bf a = frag_ld(Ub + mt * 16 * 136 + kk, 136, lane);
                v16bf b = frag_ld(Wx + n0 * 128 + kk, 128, lane);
                acc = wmma_bf16(a, b, acc);
            }
            int n = n0 + (lane & 15), mb = mt * 16 + (lane >> 4) * 8;
#pragma unroll
            for (int i = 0; i < 8; ++i) pbf[(mb + i) * 128 + n] = (bf16_t)acc[i];
        }
    __syncthreads();

    // dtA (32x32): first 24 cols of p, padded
    bf16_t* dtA = dtY;
    for (int idx = tid; idx < 32 * 32; idx += 128) {
        int l = idx >> 5, c = idx & 31;
        dtA[l * 40 + c] = (c < 24) ? pbf[l * 128 + c] : (bf16_t)0.0f;
    }
    __syncthreads();

    // GEMM3: delta = min(softplus(dtA @ sWdtPad[dir].T + sbdt), 10)
    const bf16_t* Wdt = sWdtb + dir * 128 * 32;
    bf16_t* dS = UbD;  // reuse (Ub dead)
#pragma unroll
    for (int mt = 0; mt < 2; ++mt)
#pragma unroll
        for (int j = 0; j < 2; ++j) {
            int n0 = wave * 32 + j * 16;
            v16bf a = frag_ld(dtA + mt * 16 * 40, 40, lane);
            v16bf b = frag_ld(Wdt + n0 * 32, 32, lane);
            v8f acc = {};
            acc = wmma_bf16(a, b, acc);
            int n = n0 + (lane & 15), mb = mt * 16 + (lane >> 4) * 8;
#pragma unroll
            for (int i = 0; i < 8; ++i) {
                float raw = acc[i] + sbdt[dir * 128 + n];
                float sp = raw > 20.f ? raw : log1pf(__expf(raw));
                dS[(mb + i) * 128 + n] = (bf16_t)fminf(sp, 10.f);
            }
        }
    __syncthreads();

    // sequential SSM recurrence: thread d owns all 48 states of channel d
    {
        const int d = tid;
        const float* Arow = sAlog + (dir * 128 + d) * 48;
        float Ac[48], hst[48];
#pragma unroll
        for (int n = 0; n < 48; ++n) {
            Ac[n] = -fminf(__expf(Arow[n]), 10000.f);
            hst[n] = 0.f;
        }
        const float Dp = sD[dir * 128 + d];
        bf16_t* Yb = dtY;  // reuse (dtA dead)
        for (int l = 0; l < 32; ++l) {
            float dl = (float)dS[l * 128 + d];
            float ul = uS[l * 128 + d];
            float y = 0.f;
#pragma unroll
            for (int n = 0; n < 48; ++n) {
                float dA = __expf(clampf(dl * Ac[n], -30.f, 30.f));
                float bv = (float)pbf[l * 128 + 24 + n];
                float cv = (float)pbf[l * 128 + 72 + n];
                hst[n] = dA * hst[n] + dl * bv * ul;
                y += cv * hst[n];
            }
            y += Dp * ul;
            float zv = (float)zbf[l * 128 + d];
            float gate = zv / (1.f + __expf(-zv));
            Yb[l * 136 + d] = (bf16_t)(y * gate);
        }
    }
    __syncthreads();

    // GEMM4: y @ sWout[dir].T, nan_to_num, + x residual
    const bf16_t* Wo = sWoutb + dir * 128 * 128;
    bf16_t* Yb = dtY;
    float* Yo = uS;  // reuse (u dead)
#pragma unroll
    for (int mt = 0; mt < 2; ++mt)
#pragma unroll
        for (int j = 0; j < 2; ++j) {
            int n0 = wave * 32 + j * 16;
            v8f acc = {};
#pragma unroll
            for (int kk = 0; kk < 128; kk += 32) {
                v16bf a = frag_ld(Yb + mt * 16 * 136 + kk, 136, lane);
                v16bf b = frag_ld(Wo + n0 * 128 + kk, 128, lane);
                acc = wmma_bf16(a, b, acc);
            }
            int n = n0 + (lane & 15), mb = mt * 16 + (lane >> 4) * 8;
#pragma unroll
            for (int i = 0; i < 8; ++i) {
                int m = mb + i;
                int ls = flip ? 31 - m : m;
                Yo[m * 128 + n] = nan_fix(acc[i]) + xbase[ls * 128 + n];
            }
        }
    __syncthreads();

    // LN (sng/snb) and scatter back to BHWD order (un-flip / un-transpose)
    {
        int row = tid >> 2, sub = tid & 3;  // 4 lanes per 128-wide row
        const float* r = Yo + row * 128 + sub * 32;
        float s = 0.f, s2 = 0.f, vv[32];
#pragma unroll
        for (int i = 0; i < 32; ++i) { vv[i] = r[i]; s += vv[i]; s2 += vv[i] * vv[i]; }
#pragma unroll
        for (int m = 2; m >= 1; m >>= 1) { s += __shfl_xor(s, m, 4); s2 += __shfl_xor(s2, m, 4); }
        float mean = s * (1.f / 128.f);
        float rstd = rsqrtf(s2 * (1.f / 128.f) - mean * mean + 1e-5f);
        int l = row, grow;
        if (dir < 2) {
            int w = flip ? 31 - l : l;
            grow = seq * 32 + w;                  // ((b*H+h)*W + w)
        } else {
            int hh = flip ? 31 - l : l;
            int b = seq >> 5, w = seq & 31;
            grow = (b * 32 + hh) * 32 + w;
        }
        float* dst = dir_out + (size_t)dir * 4096 * 128 + grow * 128;
#pragma unroll
        for (int i = 0; i < 32; ++i) {
            int c = sub * 32 + i;
            dst[c] = (vv[i] - mean) * rstd * sng[dir * 128 + c] + snb[dir * 128 + c];
        }
    }
}

// ---------------------------------------------------------------------------
// Kernel 4: out = grid + 0.25*sum(dirs) @ Wout.T + bout ; lnb = LN(out)
// ---------------------------------------------------------------------------
__global__ void __launch_bounds__(128) k_combine(
    const float* __restrict__ dir_out, const float* __restrict__ grid,
    const bf16_t* __restrict__ Woutb, const float* __restrict__ bout,
    const float* __restrict__ fng, const float* __restrict__ fnb,
    float* __restrict__ out, bf16_t* __restrict__ lnb) {
    __shared__ bf16_t As[16][136];
    __shared__ float outS[16][128];
    const int tile = blockIdx.x, tid = threadIdx.x;
    for (int idx = tid; idx < 16 * 128; idx += 128) {
        int r = idx >> 7, c = idx & 127;
        size_t g = (size_t)(tile * 16 + r) * 128 + c;
        float s = dir_out[g] + dir_out[g + 524288] + dir_out[g + 1048576] + dir_out[g + 1572864];
        As[r][c] = (bf16_t)(0.25f * s);
    }
    __syncthreads();
    const int wave = tid >> 5, lane = tid & 31;
#pragma unroll
    for (int j = 0; j < 2; ++j) {
        int n0 = wave * 32 + j * 16;
        v8f acc = {};
#pragma unroll
        for (int kk = 0; kk < 128; kk += 32) {
            v16bf a = frag_ld(&As[0][0] + kk, 136, lane);
            v16bf b = frag_ld(Woutb + n0 * 128 + kk, 128, lane);
            acc = wmma_bf16(a, b, acc);
        }
        int n = n0 + (lane & 15), mb = (lane >> 4) * 8;
#pragma unroll
        for (int i = 0; i < 8; ++i) {
            int m = mb + i;
            size_t g = (size_t)(tile * 16 + m) * 128 + n;
            float v = grid[g] + acc[i] + bout[n];
            out[g] = v;
            outS[m][n] = v;
        }
    }
    __syncthreads();
    int row = tid >> 3, sub = tid & 7;
    const float* r = &outS[row][sub * 16];
    float s = 0.f, s2 = 0.f, vv[16];
#pragma unroll
    for (int i = 0; i < 16; ++i) { vv[i] = r[i]; s += vv[i]; s2 += vv[i] * vv[i]; }
#pragma unroll
    for (int m = 4; m >= 1; m >>= 1) { s += __shfl_xor(s, m, 8); s2 += __shfl_xor(s2, m, 8); }
    float mean = s * (1.f / 128.f);
    float rstd = rsqrtf(s2 * (1.f / 128.f) - mean * mean + 1e-5f);
#pragma unroll
    for (int i = 0; i < 16; ++i) {
        int c = sub * 16 + i;
        lnb[(size_t)(tile * 16 + row) * 128 + c] =
            (bf16_t)((vv[i] - mean) * rstd * fng[c] + fnb[c]);
    }
}

// ---------------------------------------------------------------------------
// Kernel 5: h1 = GELU(ln @ W1.T + b1)   (M=4096, N=512, K=128)
// ---------------------------------------------------------------------------
__global__ void __launch_bounds__(128) k_mlp1(
    const bf16_t* __restrict__ lnb, const bf16_t* __restrict__ W1b,
    const float* __restrict__ b1, bf16_t* __restrict__ h1) {
    const int tile = blockIdx.x, tid = threadIdx.x, wave = tid >> 5, lane = tid & 31;
    const bf16_t* Abase = lnb + (size_t)tile * 16 * 128;
#pragma unroll
    for (int j = 0; j < 8; ++j) {
        int n0 = wave * 128 + j * 16;
        v8f acc = {};
#pragma unroll
        for (int kk = 0; kk < 128; kk += 32) {
            v16bf a = frag_ld(Abase + kk, 128, lane);
            v16bf b = frag_ld(W1b + n0 * 128 + kk, 128, lane);
            acc = wmma_bf16(a, b, acc);
        }
        int n = n0 + (lane & 15), mb = (lane >> 4) * 8;
#pragma unroll
        for (int i = 0; i < 8; ++i) {
            float t = acc[i] + b1[n];
            float g = 0.5f * t * (1.f + erff(t * 0.70710678118f));
            h1[(size_t)(tile * 16 + mb + i) * 512 + n] = (bf16_t)g;
        }
    }
}

// ---------------------------------------------------------------------------
// Kernel 6: y = nan_to_num(out + h1 @ W2.T + b2) * vm   (M=4096, N=128, K=512)
// ---------------------------------------------------------------------------
__global__ void __launch_bounds__(128) k_mlp2(
    const bf16_t* __restrict__ h1, const bf16_t* __restrict__ W2b,
    const float* __restrict__ b2, const float* __restrict__ out,
    const unsigned char* __restrict__ valid, float* __restrict__ y) {
    const int tile = blockIdx.x, tid = threadIdx.x, wave = tid >> 5, lane = tid & 31;
    const bf16_t* Abase = h1 + (size_t)tile * 16 * 512;
#pragma unroll
    for (int j = 0; j < 2; ++j) {
        int n0 = wave * 32 + j * 16;
        v8f acc = {};
#pragma unroll
        for (int kk = 0; kk < 512; kk += 32) {
            v16bf a = frag_ld(Abase + kk, 512, lane);
            v16bf b = frag_ld(W2b + n0 * 512 + kk, 512, lane);
            acc = wmma_bf16(a, b, acc);
        }
        int n = n0 + (lane & 15), mb = (lane >> 4) * 8;
#pragma unroll
        for (int i = 0; i < 8; ++i) {
            int pix = tile * 16 + mb + i;
            size_t g = (size_t)pix * 128 + n;
            float v = nan_fix(out[g] + acc[i] + b2[n]);
            y[g] = v * (valid[pix] ? 1.f : 0.f);
        }
    }
}

// ---------------------------------------------------------------------------
// Workspace layout (bytes, 256-aligned)
// ---------------------------------------------------------------------------
#define OFF_WINB   0u          // 128x128 bf16        32768
#define OFF_SWINB  32768u      // 4x256x128 bf16     262144
#define OFF_SWXB   294912u     // 4x128x128 bf16     131072 (padded)
#define OFF_SWDTB  425984u     // 4x128x32 bf16       32768 (padded)
#define OFF_SWOUTB 458752u     // 4x128x128 bf16     131072
#define OFF_WOUTB  589824u     // 128x128 bf16        32768
#define OFF_W1B    622592u     // 512x128 bf16       131072
#define OFF_W2B    753664u     // 128x512 bf16       131072
#define OFF_X      884736u     // 4096x128 f32      2097152
#define OFF_XM     2981888u    // 4096x128 f32      2097152
#define OFF_XMT    5079040u    // 4096x128 f32      2097152
#define OFF_DIR    7176192u    // 4x4096x128 f32    8388608
#define OFF_OUT    15564800u   // 4096x128 f32      2097152
#define OFF_LNB    17661952u   // 4096x128 bf16     1048576
#define OFF_H1B    18710528u   // 4096x512 bf16     4194304  -> 22904832 total

extern "C" void kernel_launch(void* const* d_in, const int* in_sizes, int n_in,
                              void* d_out, int out_size, void* d_ws, size_t ws_size,
                              hipStream_t stream) {
    (void)in_sizes; (void)n_in; (void)out_size; (void)ws_size;
    const float* grid  = (const float*)d_in[0];
    const unsigned char* valid = (const unsigned char*)d_in[1];
    const float* ng    = (const float*)d_in[2];
    const float* nb    = (const float*)d_in[3];
    const float* Win   = (const float*)d_in[4];
    const float* bin_  = (const float*)d_in[5];
    const float* convw = (const float*)d_in[6];
    const float* convb = (const float*)d_in[7];
    const float* sWin  = (const float*)d_in[8];
    const float* sWx   = (const float*)d_in[9];
    const float* sWdt  = (const float*)d_in[10];
    const float* sbdt  = (const float*)d_in[11];
    const float* sAlog = (const float*)d_in[12];
    const float* sD    = (const float*)d_in[13];
    const float* sWout = (const float*)d_in[14];
    const float* sng   = (const float*)d_in[15];
    const float* snb   = (const float*)d_in[16];
    const float* Wout  = (const float*)d_in[17];
    const float* bout  = (const float*)d_in[18];
    const float* fng   = (const float*)d_in[19];
    const float* fnb   = (const float*)d_in[20];
    const float* W1    = (const float*)d_in[21];
    const float* b1    = (const float*)d_in[22];
    const float* W2    = (const float*)d_in[23];
    const float* b2    = (const float*)d_in[24];

    char* ws = (char*)d_ws;
    bf16_t* Winb   = (bf16_t*)(ws + OFF_WINB);
    bf16_t* sWinb  = (bf16_t*)(ws + OFF_SWINB);
    bf16_t* sWxb   = (bf16_t*)(ws + OFF_SWXB);
    bf16_t* sWdtb  = (bf16_t*)(ws + OFF_SWDTB);
    bf16_t* sWoutb = (bf16_t*)(ws + OFF_SWOUTB);
    bf16_t* Woutb  = (bf16_t*)(ws + OFF_WOUTB);
    bf16_t* W1b    = (bf16_t*)(ws + OFF_W1B);
    bf16_t* W2b    = (bf16_t*)(ws + OFF_W2B);
    float*  x_ws   = (float*)(ws + OFF_X);
    float*  xm_ws  = (float*)(ws + OFF_XM);
    float*  xmT_ws = (float*)(ws + OFF_XMT);
    float*  dir_ws = (float*)(ws + OFF_DIR);
    float*  out_ws = (float*)(ws + OFF_OUT);
    bf16_t* lnb_ws = (bf16_t*)(ws + OFF_LNB);
    bf16_t* h1_ws  = (bf16_t*)(ws + OFF_H1B);
    float*  y_out  = (float*)d_out;

    // weight prep (bf16 copies + padded layouts)
    k_cvt<<<64, 256, 0, stream>>>(Win, Winb, 128 * 128);
    k_cvt<<<512, 256, 0, stream>>>(sWin, sWinb, 4 * 256 * 128);
    k_cvt<<<256, 256, 0, stream>>>(sWout, sWoutb, 4 * 128 * 128);
    k_cvt<<<64, 256, 0, stream>>>(Wout, Woutb, 128 * 128);
    k_cvt<<<256, 256, 0, stream>>>(W1, W1b, 512 * 128);
    k_cvt<<<256, 256, 0, stream>>>(W2, W2b, 128 * 512);
    k_padwx<<<256, 256, 0, stream>>>(sWx, sWxb);
    k_padwdt<<<64, 256, 0, stream>>>(sWdt, sWdtb);

    k_ln_proj<<<256, 128, 0, stream>>>(grid, ng, nb, Winb, bin_, x_ws);
    k_conv<<<4096, 128, 0, stream>>>(x_ws, convw, convb, valid, xm_ws, xmT_ws);
    k_scan<<<512, 128, 0, stream>>>(xm_ws, xmT_ws, sWinb, sWxb, sWdtb, sbdt, sAlog,
                                    sD, sWoutb, sng, snb, dir_ws);
    k_combine<<<256, 128, 0, stream>>>(dir_ws, grid, Woutb, bout, fng, fnb, out_ws,
                                       lnb_ws);
    k_mlp1<<<256, 128, 0, stream>>>(lnb_ws, W1b, b1, h1_ws);
    k_mlp2<<<256, 128, 0, stream>>>(h1_ws, W2b, b2, out_ws, valid, y_out);
}
